// AttnBlock_84189948936313
// MI455X (gfx1250) — matrix operand order repaired and checked by reading, not compiled, not run on hardware
//
#include <hip/hip_runtime.h>
#include <hip/hip_bf16.h>

// ---------------------------------------------------------------------------
// CDNA5 (gfx1250, wave32) WMMA attention block.
// All GEMM fragments are contiguous 16/32-byte vector loads by construction:
//   hT,qT,kT,h2T stored [b][t][c];  v stored [b][c][t];  P stored [b][k][q].
// ---------------------------------------------------------------------------
typedef __attribute__((ext_vector_type(16))) __bf16 v16bf;
typedef __attribute__((ext_vector_type(8)))  __bf16 v8bf;
typedef __attribute__((ext_vector_type(8)))  float  v8f;
typedef __attribute__((ext_vector_type(4)))  unsigned int u32x4;
typedef __attribute__((ext_vector_type(8)))  int i32x8;
typedef __attribute__((ext_vector_type(4)))  int i32x4;

#define BATCH 4
#define CCH   256
#define TT    4096
#define GRP   32
#define SROW_LD (TT + 4)   // LDS row pad: 16 distinct banks for column reads

__device__ __forceinline__ __bf16 f2bf(float f) {
    unsigned u = __builtin_bit_cast(unsigned, f);
    unsigned r = (u + 0x7FFFu + ((u >> 16) & 1u)) >> 16;
    unsigned short s = (unsigned short)r;
    return __builtin_bit_cast(__bf16, s);
}

__device__ __forceinline__ v8f wmma_bf16(v16bf a, v16bf b, v8f c) {
    return __builtin_amdgcn_wmma_f32_16x16x32_bf16(false, a, false, b, (short)0, c,
                                                   false, false);
}

// A-fragment (16x32 MxK) from row-major [m][k], stride in elements.
// lane<16: m=lane, K={0..15}; lane>=16: m=lane-16, K={16..31}
__device__ __forceinline__ v16bf load_a_rm(const __bf16* base, int stride, int lane) {
    int m  = lane & 15;
    int kb = (lane & 16) ? 16 : 0;
    const __bf16* p = base + (size_t)m * stride + kb;
    v8bf lo = *(const v8bf*)(p);
    v8bf hi = *(const v8bf*)(p + 8);
    return __builtin_shufflevector(lo, hi, 0,1,2,3,4,5,6,7,8,9,10,11,12,13,14,15);
}

// B-fragment (32x16 KxN) from K-contiguous [n][k] memory (stride elems/row):
// lane n = lane&15, kbase = lane<16 ? 0 : 16, elements K=kbase..kbase+15.
// Single 32B read per lane.
__device__ __forceinline__ v16bf load_b_klast(const __bf16* base, int stride, int lane) {
    int n  = lane & 15;
    int kb = (lane & 16) ? 16 : 0;
    return *(const v16bf*)(base + (size_t)n * stride + kb);
}

// ---------------------------------------------------------------------------
// Kernel 1: GroupNorm -> hT[b][t][c] bf16
// ---------------------------------------------------------------------------
__global__ void gn_kernel(const float* __restrict__ x, const float* __restrict__ gsc,
                          const float* __restrict__ gbi, __bf16* __restrict__ hT) {
    const int b  = blockIdx.x >> 5;
    const int g  = blockIdx.x & 31;
    const int c0 = g * (CCH / GRP);
    const size_t base = ((size_t)b * CCH + c0) * TT;
    const int tid = threadIdx.x;
    const int N = (CCH / GRP) * TT;                 // 32768

    float s = 0.f, ss = 0.f;
    for (int i = tid; i < N; i += 256) {
        float v = x[base + (size_t)(i >> 12) * TT + (i & (TT - 1))];
        s += v; ss += v * v;
    }
    __shared__ float r1[256], r2[256];
    r1[tid] = s; r2[tid] = ss;
    __syncthreads();
    for (int off = 128; off; off >>= 1) {
        if (tid < off) { r1[tid] += r1[tid + off]; r2[tid] += r2[tid + off]; }
        __syncthreads();
    }
    const float mean = r1[0] * (1.0f / N);
    const float var  = r2[0] * (1.0f / N) - mean * mean;
    const float rstd = rsqrtf(var + 1e-6f);

    for (int i = tid; i < N; i += 256) {
        int cc = i >> 12, t = i & (TT - 1);
        float v = x[base + (size_t)cc * TT + t];
        float o = (v - mean) * rstd * gsc[c0 + cc] + gbi[c0 + cc];
        hT[((size_t)b * TT + t) * CCH + c0 + cc] = f2bf(o);
    }
}

// ---------------------------------------------------------------------------
// Kernel 2/5: out[b,o,t] = sum_c W[o,c] * BmT[b,t,c] + bias[o]  (+ residual)
// grid (T/64, C/64, B), 128 threads (4 waves).  W staged once in LDS (32 KB),
// A-fragments hoisted; B-fragments are single 32 B global loads from t-major.
// TRANSP: bf16 out [b][t][o];  else bf16 [b][o][t];  RESID: fp32 x+gemm+bias.
// ---------------------------------------------------------------------------
template <bool TRANSP, bool RESID>
__global__ void cproj_kernel(const float* __restrict__ W, const float* __restrict__ bias,
                             const __bf16* __restrict__ BmT, __bf16* __restrict__ outb,
                             float* __restrict__ outf, const float* __restrict__ xres) {
    __shared__ __align__(32) __bf16 Abuf[64 * 256];   // [m][k], whole K
    const int b    = blockIdx.z;
    const int n0   = blockIdx.x * 64;
    const int m0   = blockIdx.y * 64;
    const int tid  = threadIdx.x;
    const int lane = tid & 31;
    const int wave = tid >> 5;

    // stage A (64x256 fp32 -> bf16), coalesced along k
    for (int it = 0; it < 128; ++it) {
        int i = tid + it * 128;
        int m = i >> 8, k = i & 255;
        Abuf[m * 256 + k] = f2bf(W[(size_t)(m0 + m) * CCH + k]);
    }
    __syncthreads();

    v16bf afr[8];
#pragma unroll
    for (int cc = 0; cc < 8; ++cc)
        afr[cc] = load_a_rm(Abuf + (wave * 16) * 256 + cc * 32, 256, lane);

    const __bf16* brow = BmT + ((size_t)b * TT + n0) * CCH;
    v8f acc[4] = {};
#pragma unroll
    for (int cc = 0; cc < 8; ++cc) {
#pragma unroll
        for (int j = 0; j < 4; ++j) {
            v16bf bf = load_b_klast(brow + (size_t)(j * 16) * CCH + cc * 32, CCH, lane);
            acc[j] = wmma_bf16(afr[cc], bf, acc[j]);
        }
    }

    const int n  = lane & 15;
    const int mh = (lane & 16) ? 8 : 0;
#pragma unroll
    for (int j = 0; j < 4; ++j) {
#pragma unroll
        for (int r = 0; r < 8; ++r) {
            int row = m0 + wave * 16 + r + mh;
            int col = n0 + j * 16 + n;
            float val = acc[j][r] + bias[row];
            if (RESID) {
                size_t idx = ((size_t)b * CCH + row) * TT + col;
                outf[idx] = xres[idx] + val;
            } else if (TRANSP) {
                outb[((size_t)b * TT + col) * CCH + row] = f2bf(val);
            } else {
                outb[((size_t)b * CCH + row) * TT + col] = f2bf(val);
            }
        }
    }
}

// ---------------------------------------------------------------------------
// Kernel 3: S = (Q^T K)/16 for 16 q rows; full 16x4096 row-block in LDS
// (CDNA5 320KB WGP LDS); softmax over k; emit P transposed [b][k][q] bf16.
// Q-tile (16x256, contiguous 8KB in qT) fetched via Tensor Data Mover.
// grid (T/16, B), 256 threads (8 waves).
// ---------------------------------------------------------------------------
__global__ void attn_softmax_kernel(const __bf16* __restrict__ qT,
                                    const __bf16* __restrict__ kT,
                                    __bf16* __restrict__ PT) {
    extern __shared__ __align__(32) char smem_raw[];
    float*  Srow  = (float*)smem_raw;                                   // [16][SROW_LD]
    __bf16* Atile = (__bf16*)(smem_raw + 16 * SROW_LD * 4);             // [16][256]
    float*  mArr  = (float*)(smem_raw + 16 * SROW_LD * 4 + 16 * CCH * 2);
    float*  lArr  = mArr + 16;

    const int b    = blockIdx.y;
    const int q0   = blockIdx.x * 16;
    const int tid  = threadIdx.x;
    const int lane = tid & 31;
    const int wave = tid >> 5;
    const __bf16* qsrc = qT + ((size_t)b * TT + q0) * CCH;

#if __has_builtin(__builtin_amdgcn_tensor_load_to_lds)
    // --- TDM: 2D tensor [T][C] bf16, tile 16 x 256, into Atile -------------
    if (wave == 0) {
        unsigned lds_off = (unsigned)(size_t)(void*)Atile;
        unsigned long long ga = (unsigned long long)(size_t)(const void*)qsrc;
        u32x4 g0;
        g0[0] = 1u;                                           // count=1, user mode
        g0[1] = lds_off;                                      // lds_addr
        g0[2] = (unsigned)(ga & 0xFFFFFFFFu);                 // global_addr[31:0]
        g0[3] = (unsigned)((ga >> 32) & 0x1FFFFFFu) | (2u << 30);  // [56:32] | type=2
        i32x8 g1;
        g1[0] = (int)(1u << 16);                              // data_size=1 (2B)
        g1[1] = (int)((CCH & 0xFFFFu) << 16);                 // tensor_dim0 lo16
        g1[2] = (int)(((CCH >> 16) & 0xFFFFu) | ((TT & 0xFFFFu) << 16)); // d0 hi|d1 lo
        g1[3] = (int)(((TT >> 16) & 0xFFFFu) | ((unsigned)CCH << 16));   // d1 hi|tile_dim0
        g1[4] = 16;                                           // tile_dim1=16, tile_dim2=0
        g1[5] = CCH;                                          // tensor_dim0_stride lo32
        g1[6] = 0;                                            // d0s hi16 | d1s lo16
        g1[7] = 0;                                            // d1s hi32
        i32x4 z4 = {0, 0, 0, 0};
        i32x8 z8 = {0, 0, 0, 0, 0, 0, 0, 0};
        __builtin_amdgcn_tensor_load_to_lds(g0, g1, z4, z4, z8, 0);
        __builtin_amdgcn_s_wait_tensorcnt(0);
    }
    __syncthreads();
#else
    {
        const v8bf* src = (const v8bf*)qsrc;
        v8bf* dst = (v8bf*)Atile;
        for (int i = tid; i < 16 * CCH / 8; i += 256) dst[i] = src[i];
        __syncthreads();
    }
#endif

    // Hoist all 8 A-fragments (K=256 in 32-steps)
    v16bf afr[8];
#pragma unroll
    for (int cc = 0; cc < 8; ++cc)
        afr[cc] = load_a_rm(Atile + cc * 32, CCH, lane);

    // Each wave sweeps k-tiles jt = wave, wave+8, ...; B frags contiguous in kT
    for (int jt = wave; jt < TT / 16; jt += 8) {
        const int n0 = jt * 16;
        const __bf16* brow = kT + ((size_t)b * TT + n0) * CCH;
        v8f acc = {};
#pragma unroll
        for (int cc = 0; cc < 8; ++cc) {
            v16bf bf = load_b_klast(brow + cc * 32, CCH, lane);
            acc = wmma_bf16(afr[cc], bf, acc);
        }
        const int n  = lane & 15;
        const int mh = (lane & 16) ? 8 : 0;
#pragma unroll
        for (int r = 0; r < 8; ++r)
            Srow[(size_t)(r + mh) * SROW_LD + n0 + n] = acc[r] * 0.0625f;  // C^-0.5
    }
    __syncthreads();

    // Softmax stats per row; wave w owns rows 2w, 2w+1
    for (int rr = 0; rr < 2; ++rr) {
        const int row = wave * 2 + rr;
        float* Sr = Srow + (size_t)row * SROW_LD;
        float mx = -3.4e38f;
        for (int i = lane; i < TT; i += 32) mx = fmaxf(mx, Sr[i]);
#pragma unroll
        for (int off = 16; off; off >>= 1) mx = fmaxf(mx, __shfl_xor(mx, off, 32));
        float sum = 0.f;
        for (int i = lane; i < TT; i += 32) sum += __expf(Sr[i] - mx);
#pragma unroll
        for (int off = 16; off; off >>= 1) sum += __shfl_xor(sum, off, 32);
        if (lane == 0) { mArr[row] = mx; lArr[row] = 1.0f / sum; }
    }
    __syncthreads();

    // Write P transposed [b][k][q0+qi]; lanes sweep qi fast (32B segments),
    // padded Srow stride keeps the 16 column reads on distinct banks.
    for (int i = tid; i < 16 * TT; i += 256) {
        int qi = i & 15, k = i >> 4;
        float p = __expf(Srow[(size_t)qi * SROW_LD + k] - mArr[qi]) * lArr[qi];
        PT[((size_t)b * TT + k) * TT + q0 + qi] = f2bf(p);
    }
}

// ---------------------------------------------------------------------------
// Kernel 4: h2[b,c,k] = sum_q v[b,c,q] * P[b,q,k];  A from v [c][q] (contig),
// B from PT [k][q] (contig).  Output h2T [b][t=k][c].
// grid (T/64, C/64, B), 128 threads (4 waves).
// ---------------------------------------------------------------------------
__global__ void av_gemm_kernel(const __bf16* __restrict__ vb,
                               const __bf16* __restrict__ PT,
                               __bf16* __restrict__ h2T) {
    const int b    = blockIdx.z;
    const int n0   = blockIdx.x * 64;   // k-column tile
    const int m0   = blockIdx.y * 64;   // c tile
    const int lane = threadIdx.x & 31;
    const int wave = threadIdx.x >> 5;
    const size_t cb = (size_t)b * CCH * TT;
    const size_t pb = (size_t)b * TT * TT;

    const __bf16* arow = vb + cb + (size_t)(m0 + wave * 16) * TT;
    const __bf16* prow = PT + pb + (size_t)n0 * TT;
    v8f acc[4] = {};
    for (int k0 = 0; k0 < TT; k0 += 32) {
        __builtin_prefetch(arow + k0 + 128, 0, 0);   // global_prefetch_b8
        v16bf a = load_a_rm(arow + k0, TT, lane);
#pragma unroll
        for (int j = 0; j < 4; ++j) {
            v16bf bf = load_b_klast(prow + (size_t)(j * 16) * TT + k0, TT, lane);
            acc[j] = wmma_bf16(a, bf, acc[j]);
        }
    }
    const int n  = lane & 15;
    const int mh = (lane & 16) ? 8 : 0;
#pragma unroll
    for (int j = 0; j < 4; ++j) {
#pragma unroll
        for (int r = 0; r < 8; ++r) {
            int row = m0 + wave * 16 + r + mh;      // c
            int col = n0 + j * 16 + n;              // k
            h2T[((size_t)b * TT + col) * CCH + row] = f2bf(acc[j][r]);
        }
    }
}

// ---------------------------------------------------------------------------
// Launch.  Workspace: 5 * B*T*C bf16 (hT,qT,kT,v,h2T) + B*T*T bf16 (PT).
// ---------------------------------------------------------------------------
extern "C" void kernel_launch(void* const* d_in, const int* in_sizes, int n_in,
                              void* d_out, int out_size, void* d_ws, size_t ws_size,
                              hipStream_t stream) {
    const float* x   = (const float*)d_in[0];
    const float* gsc = (const float*)d_in[1];
    const float* gbi = (const float*)d_in[2];
    const float* wq  = (const float*)d_in[3];
    const float* bq  = (const float*)d_in[4];
    const float* wk  = (const float*)d_in[5];
    const float* bk  = (const float*)d_in[6];
    const float* wv  = (const float*)d_in[7];
    const float* bv  = (const float*)d_in[8];
    const float* wo  = (const float*)d_in[9];
    const float* bo  = (const float*)d_in[10];
    float* out = (float*)d_out;

    const size_t NSZ = (size_t)BATCH * CCH * TT;
    __bf16* hT  = (__bf16*)d_ws;
    __bf16* qTp = hT + NSZ;
    __bf16* kTp = qTp + NSZ;
    __bf16* vb  = kTp + NSZ;
    __bf16* h2T = vb + NSZ;
    __bf16* PT  = h2T + NSZ;                          // B*T*T elements

    gn_kernel<<<dim3(BATCH * GRP), dim3(256), 0, stream>>>(x, gsc, gbi, hT);

    dim3 gg(TT / 64, CCH / 64, BATCH), gb(128);
    cproj_kernel<true,  false><<<gg, gb, 0, stream>>>(wq, bq, hT, qTp, nullptr, nullptr);
    cproj_kernel<true,  false><<<gg, gb, 0, stream>>>(wk, bk, hT, kTp, nullptr, nullptr);
    cproj_kernel<false, false><<<gg, gb, 0, stream>>>(wv, bv, hT, vb,  nullptr, nullptr);

    size_t smem3 = (size_t)16 * SROW_LD * 4 + (size_t)16 * CCH * 2 + 2 * 16 * 4;
    attn_softmax_kernel<<<dim3(TT / 16, BATCH), dim3(256), smem3, stream>>>(qTp, kTp, PT);

    av_gemm_kernel<<<gg, gb, 0, stream>>>(vb, PT, h2T);

    cproj_kernel<false, true><<<gg, gb, 0, stream>>>(wo, bo, h2T, nullptr, out, x);
}
